// GraphAttentionLayer_23416161698232
// MI455X (gfx1250) — matrix operand order repaired and checked by reading, not compiled, not run on hardware
//
#include <hip/hip_runtime.h>
#include <hip/hip_bf16.h>

// ---- CDNA5 WMMA fragment types (wave32) ----
typedef __attribute__((ext_vector_type(16))) __bf16 v16bf;  // A/B fragment: 16x32 / 32x16 bf16
typedef __attribute__((ext_vector_type(8)))  __bf16 v8bf;   // half fragment (16 bytes)
typedef __attribute__((ext_vector_type(8)))  float  v8f;    // C/D fragment: 16x16 f32

// Problem constants (from reference setup_inputs)
#define GB    8      // batch
#define GN    2048   // nodes
#define GFIN  512
#define GFOUT 256
#define GALPHA 0.2f

// CDNA5 async global->LDS path (gfx1250): gate on builtin availability so the
// sync fallback keeps the TU compiling on host pass / older toolchains.
#if defined(__AMDGCN__) && __has_builtin(__builtin_amdgcn_global_load_async_to_lds_b128)
#define USE_ASYNC_LDS 1
#else
#define USE_ASYNC_LDS 0
#endif

#if USE_ASYNC_LDS
// Builtin signature (from hipcc diagnostic): param 1 is a GLOBAL (addrspace(1))
// pointer to an int vector-of-4; LDS side is addrspace(3).
typedef int v4i __attribute__((vector_size(16)));
typedef __attribute__((address_space(1))) v4i v4i_g;  // global 128-bit chunk
typedef __attribute__((address_space(3))) v4i v4i_l;  // LDS    128-bit chunk
#endif

__device__ __forceinline__ float leakyf(float x) { return x > 0.f ? x : GALPHA * x; }

__device__ __forceinline__ v8f wmma_bf16(v16bf a, v16bf b, v8f c) {
  // D = A(16x32) * B(32x16) + C ; v_wmma_f32_16x16x32_bf16
  return __builtin_amdgcn_wmma_f32_16x16x32_bf16(false, a, false, b, (short)0, c, false, false);
}

#define SHUF16(lo, hi) \
  __builtin_shufflevector((lo), (hi), 0, 1, 2, 3, 4, 5, 6, 7, 8, 9, 10, 11, 12, 13, 14, 15)

// A fragment (16x32 bf16, MxK): lane = row m (lane%16).
// lane<16 holds K = {k0+0..7, k0+16..23}; lane>=16 holds K = {k0+8..15, k0+24..31}.
__device__ __forceinline__ v16bf load_a_frag_bf(const __bf16* __restrict__ rowp, int k0, int hi) {
  const v8bf c0 = *(const v8bf*)(rowp + k0 + 8 * hi);
  const v8bf c1 = *(const v8bf*)(rowp + k0 + 16 + 8 * hi);
  return SHUF16(c0, c1);
}

// B fragment (32x16 bf16, KxN): lane = col n (lane%16); K = {k0+8*hi+0..7, k0+16+8*hi+0..7}.
__device__ __forceinline__ v16bf load_b_frag_bf(const __bf16* __restrict__ colp, int k0, int hi) {
  const v8bf c0 = *(const v8bf*)(colp + k0 + 8 * hi);
  const v8bf c1 = *(const v8bf*)(colp + k0 + 16 + 8 * hi);
  return SHUF16(c0, c1);
}

// -------------------------------------------------------------------------
// Kernel 0: one-shot f32 -> bf16 conversion of h and W (8 elems/thread).
// Removes all v_cvt from the GEMM inner loop and halves its global traffic.
// -------------------------------------------------------------------------
#define H_VEC8 ((GB * GN * GFIN) / 8)   // 1048576
#define W_VEC8 ((GFOUT * GFIN) / 8)     // 16384

__global__ __launch_bounds__(256) void gat_cvt(const float* __restrict__ h,
                                               const float* __restrict__ W,
                                               __bf16* __restrict__ hb,
                                               __bf16* __restrict__ wb) {
  const int g = blockIdx.x * 256 + threadIdx.x;  // < H_VEC8 + W_VEC8 exactly
  const float* src;
  __bf16* dst;
  if (g < H_VEC8) {
    src = h + (size_t)g * 8;
    dst = hb + (size_t)g * 8;
  } else {
    const int gw = g - H_VEC8;
    src = W + (size_t)gw * 8;
    dst = wb + (size_t)gw * 8;
  }
  const float4 f0 = *(const float4*)(src);
  const float4 f1 = *(const float4*)(src + 4);
  v8bf o;
  o[0] = (__bf16)f0.x; o[1] = (__bf16)f0.y; o[2] = (__bf16)f0.z; o[3] = (__bf16)f0.w;
  o[4] = (__bf16)f1.x; o[5] = (__bf16)f1.y; o[6] = (__bf16)f1.z; o[7] = (__bf16)f1.w;
  *(v8bf*)dst = o;
}

// -------------------------------------------------------------------------
// Kernel 1: Wh = h @ W^T, written TRANSPOSED as bf16: whT[b][o][i]
// Grid: (M/128, FOUT/64) = (128, 4), block 256 (8 waves).
// Wave (wy,wx) owns a 32x32 output tile = 2x2 WMMA fragments.
// -------------------------------------------------------------------------
__global__ __launch_bounds__(256) void gat_gemm1(const __bf16* __restrict__ hb,
                                                 const __bf16* __restrict__ wb,
                                                 __bf16* __restrict__ whT) {
  const int tid  = threadIdx.x;
  const int lane = tid & 31;
  const int wv   = tid >> 5;     // 0..7
  const int wy   = wv & 3;       // row 32-block in WG
  const int wx   = wv >> 2;      // col 32-block in WG (0..1)
  const int ln   = lane & 15;
  const int hi   = lane >> 4;

  const int row_base = blockIdx.x * 128 + wy * 32;  // flat row in [0, B*N)
  const int col_base = blockIdx.y * 64 + wx * 32;   // o in [0, FOUT)

  const __bf16* a_row0 = hb + (size_t)(row_base + ln) * GFIN;
  const __bf16* a_row1 = hb + (size_t)(row_base + 16 + ln) * GFIN;
  const __bf16* b_col0 = wb + (size_t)(col_base + ln) * GFIN;      // W[o][:] contiguous
  const __bf16* b_col1 = wb + (size_t)(col_base + 16 + ln) * GFIN;

  v8f c00 = {}, c01 = {}, c10 = {}, c11 = {};

  for (int k0 = 0; k0 < GFIN; k0 += 32) {
    const v16bf a0 = load_a_frag_bf(a_row0, k0, hi);
    const v16bf a1 = load_a_frag_bf(a_row1, k0, hi);
    const v16bf b0 = load_b_frag_bf(b_col0, k0, hi);
    const v16bf b1 = load_b_frag_bf(b_col1, k0, hi);
    c00 = wmma_bf16(a0, b0, c00);
    c01 = wmma_bf16(a0, b1, c01);
    c10 = wmma_bf16(a1, b0, c10);
    c11 = wmma_bf16(a1, b1, c11);
  }

  // C layout: lane holds col n=ln, rows m = r + 8*hi (r=0..7) -> 8 consecutive
  // global rows => one packed 16B bf16 store per fragment into whT[b][o][i].
#pragma unroll
  for (int fm = 0; fm < 2; ++fm) {
#pragma unroll
    for (int fn = 0; fn < 2; ++fn) {
      const v8f& c = fm == 0 ? (fn == 0 ? c00 : c01) : (fn == 0 ? c10 : c11);
      const int grow = row_base + fm * 16 + 8 * hi;  // first of 8 consecutive rows
      const int b    = grow >> 11;                    // / GN
      const int i    = grow & (GN - 1);
      const int o    = col_base + fn * 16 + ln;
      v8bf st;
#pragma unroll
      for (int r = 0; r < 8; ++r) st[r] = (__bf16)c[r];
      *(v8bf*)(whT + (((size_t)(b * GFOUT + o)) << 11) + i) = st;
    }
  }
}

// -------------------------------------------------------------------------
// Kernel 2: s1[b,i] = sum_o Wh[b,i,o]*a1[o]; s2 likewise. Reads whT columns.
// -------------------------------------------------------------------------
__global__ __launch_bounds__(256) void gat_scores(const __bf16* __restrict__ whT,
                                                  const float* __restrict__ a1,
                                                  const float* __restrict__ a2,
                                                  float* __restrict__ s1,
                                                  float* __restrict__ s2) {
  const int g = blockIdx.x * 256 + threadIdx.x;  // flat (b,i)
  const int b = g >> 11;
  const int i = g & (GN - 1);
  const __bf16* col = whT + (((size_t)b * GFOUT) << 11) + i;
  float acc1 = 0.f, acc2 = 0.f;
#pragma unroll 4
  for (int o = 0; o < GFOUT; ++o) {
    const float w = (float)col[(size_t)o << 11];
    acc1 += w * a1[o];
    acc2 += w * a2[o];
  }
  s1[g] = acc1;
  s2[g] = acc2;
}

// -------------------------------------------------------------------------
// Kernel 3: streaming softmax-attention + P @ Wh + ELU, double-buffered LDS.
//   e_ij = leaky(s1_i + s2_j); leaky is monotone, so the row max is exactly
//   leaky(s1_i + max_j s2_j): one pass, no online rescaling.
// Grid: (N/64, B), block 128 (4 waves); each wave owns 16 rows x 256 cols.
// -------------------------------------------------------------------------
#define TSTRIDE 80  // bytes per o-row in LDS tile (32 bf16 + 8 pad) -> conflict-free
#define TBYTES (GFOUT * TSTRIDE)

__device__ __forceinline__ void stage_tile(char* __restrict__ tilep,
                                           const __bf16* __restrict__ wbase,
                                           int j0, int tid) {
#pragma unroll
  for (int rr = 0; rr < 2; ++rr) {
    const int o = tid + rr * 128;
    const char* g = (const char*)(wbase + (((size_t)o) << 11) + j0);  // 64B row chunk
    char* l = tilep + o * TSTRIDE;
#if USE_ASYNC_LDS
#pragma unroll
    for (int c = 0; c < 4; ++c) {
      __builtin_amdgcn_global_load_async_to_lds_b128(
          (v4i_g*)(g + c * 16), (v4i_l*)(l + c * 16), 0, 0);
    }
#else
    const float4* s = (const float4*)g;
    float4* d = (float4*)l;
    d[0] = s[0]; d[1] = s[1]; d[2] = s[2]; d[3] = s[3];
#endif
  }
}

__device__ __forceinline__ void commit_stage() {
#if USE_ASYNC_LDS
#if __has_builtin(__builtin_amdgcn_s_wait_asynccnt)
  __builtin_amdgcn_s_wait_asynccnt(0);
#else
  asm volatile("s_wait_asynccnt 0x0" ::: "memory");
#endif
#endif
  __syncthreads();
}

__global__ __launch_bounds__(128) void gat_attn(const __bf16* __restrict__ whT,
                                                const float* __restrict__ s1,
                                                const float* __restrict__ s2,
                                                float* __restrict__ out) {
  __shared__ __align__(16) char tileA[TBYTES];  // 20 KB
  __shared__ __align__(16) char tileB[TBYTES];  // 20 KB (double buffer)
  __shared__ float red[128];

  const int tid  = threadIdx.x;
  const int lane = tid & 31;
  const int wv   = tid >> 5;   // 0..3
  const int ln   = lane & 15;
  const int hi   = lane >> 4;
  const int b    = blockIdx.y;

  const float* s1b = s1 + ((size_t)b << 11);
  const float* s2b = s2 + ((size_t)b << 11);
  const __bf16* wbase = whT + (((size_t)b * GFOUT) << 11);

  // --- cooperative max_j s2[b][j] ---
  float lm = -3.0e38f;
  for (int j = tid; j < GN; j += 128) lm = fmaxf(lm, s2b[j]);
  red[tid] = lm;
  __syncthreads();
  for (int s = 64; s > 0; s >>= 1) {
    if (tid < s) red[tid] = fmaxf(red[tid], red[tid + s]);
    __syncthreads();
  }
  const float s2max = red[0];
  __syncthreads();

  const int i_row = blockIdx.x * 64 + wv * 16 + ln;  // row this lane feeds into A frags
  const float s1row = s1b[i_row];
  const float mrow  = leakyf(s1row + s2max);         // exact row max of e

  v8f acc[16];
#pragma unroll
  for (int fn = 0; fn < 16; ++fn) acc[fn] = (v8f){};
  float lsum = 0.f;

  // ping-pong buffers selected by runtime ternary (no pointer array: clang
  // would constant-fold it into an addrspacecast static initializer that
  // ld.lld cannot relocate).
  bool pb = false;  // false -> current tile is tileA
  stage_tile(tileA, wbase, 0, tid);
  commit_stage();

  for (int j0 = 0; j0 < GN; j0 += 32) {
    char* cur = pb ? tileB : tileA;
    char* nxt = pb ? tileA : tileB;
    // prefetch next tile into the other buffer while we compute this one
    if (j0 + 32 < GN) stage_tile(nxt, wbase, j0 + 32, tid);

    // --- build P A-fragment in registers: p = exp(leaky(s1_i + s2_j) - m) ---
    const float4 q0 = *(const float4*)(s2b + j0 + 8 * hi);
    const float4 q1 = *(const float4*)(s2b + j0 + 8 * hi + 4);
    const float4 q2 = *(const float4*)(s2b + j0 + 16 + 8 * hi);
    const float4 q3 = *(const float4*)(s2b + j0 + 16 + 8 * hi + 4);
    const float sv[16] = {q0.x, q0.y, q0.z, q0.w, q1.x, q1.y, q1.z, q1.w,
                          q2.x, q2.y, q2.z, q2.w, q3.x, q3.y, q3.z, q3.w};
    v16bf a;
#pragma unroll
    for (int e = 0; e < 16; ++e) {
      const float pv = __expf(leakyf(s1row + sv[e]) - mrow);
      lsum += pv;
      a[e] = (__bf16)pv;
    }

    // --- 16 column fragments: acc[fn] += P(16x32) * Wh(32x16) ---
#pragma unroll
    for (int fn = 0; fn < 16; ++fn) {
      const char* srcp = cur + (fn * 16 + ln) * TSTRIDE + hi * 16;
      const v8bf lo = *(const v8bf*)srcp;
      const v8bf hb = *(const v8bf*)(srcp + 32);
      const v16bf bf = SHUF16(lo, hb);
      acc[fn] = wmma_bf16(a, bf, acc[fn]);
    }
    commit_stage();  // next buffer resident + all waves done reading this one
    pb = !pb;
  }

  // --- combine the two j-halves of each row's denominator (lanes m, m+16) ---
  lsum += __shfl_xor(lsum, 16, 32);
  float linv[8];
#pragma unroll
  for (int r = 0; r < 8; ++r) {
    const float l = __shfl(lsum, r + 8 * hi, 32);  // row m = r + 8*hi lives in lane m
    linv[r] = 1.0f / l;
  }

  // --- normalize, ELU, store f32 out[b][i][o] ---
  const int i0 = blockIdx.x * 64 + wv * 16 + 8 * hi;
#pragma unroll
  for (int fn = 0; fn < 16; ++fn) {
    const int o = fn * 16 + ln;
#pragma unroll
    for (int r = 0; r < 8; ++r) {
      float v = acc[fn][r] * linv[r];
      v = v > 0.f ? v : (__expf(v) - 1.f);
      out[((((size_t)b << 11) + (i0 + r)) << 8) + o] = v;
    }
  }
}

// -------------------------------------------------------------------------
extern "C" void kernel_launch(void* const* d_in, const int* in_sizes, int n_in,
                              void* d_out, int out_size, void* d_ws, size_t ws_size,
                              hipStream_t stream) {
  (void)in_sizes; (void)n_in; (void)out_size; (void)ws_size;
  const float* h  = (const float*)d_in[0];  // [8,2048,512]
  const float* W  = (const float*)d_in[1];  // [256,512]
  const float* a1 = (const float*)d_in[2];  // [1,256]
  const float* a2 = (const float*)d_in[3];  // [1,256]
  float* out = (float*)d_out;               // [8,2048,256]

  char* ws = (char*)d_ws;
  __bf16* whT = (__bf16*)ws;                                     // 8 MB: [8][256][2048]
  size_t off = (size_t)GB * GFOUT * GN * 2;
  float* s1 = (float*)(ws + off);                                // 64 KB
  float* s2 = s1 + (size_t)GB * GN;                              // 64 KB
  off += (size_t)GB * GN * 4 * 2;
  __bf16* hb = (__bf16*)(ws + off);                              // 16 MB: h in bf16
  off += (size_t)GB * GN * GFIN * 2;
  __bf16* wb = (__bf16*)(ws + off);                              // 256 KB: W in bf16

  gat_cvt<<<(H_VEC8 + W_VEC8) / 256, 256, 0, stream>>>(h, W, hb, wb);

  dim3 gA((GB * GN) / 128, GFOUT / 64);
  gat_gemm1<<<gA, 256, 0, stream>>>(hb, wb, whT);

  gat_scores<<<(GB * GN) / 256, 256, 0, stream>>>(whT, a1, a2, s1, s2);

  dim3 gC(GN / 64, GB);
  gat_attn<<<gC, 128, 0, stream>>>(whT, s1, s2, out);
}
